// Net_1735166787999
// MI455X (gfx1250) — matrix operand order repaired and checked by reading, not compiled, not run on hardware
//
#include <hip/hip_runtime.h>
#include <hip/hip_bf16.h>
#include <stdint.h>

// ---------------------------------------------------------------------------
// MI455X / gfx1250: wave32, WMMA 16x16x32 f16 -> f32 accumulate.
// Pipeline (only the z_mu branch is live in the reference):
//   x(16384x3) -> fc1(3,64,64,128) -> mu_nn(128,128,128,256) = hm (f16)
//   per-node segment_max over sorted-by-dst edge list (binary search, no atomics)
//   concat pos -> gmu(259->256->512->1024) -> per-graph max (8x1024)
//   head1(1024->256) -> head2(256->128) -> out(128->3)  => d_out (24 f32)
// GEMMs: v_wmma_f32_16x16x32_f16 + double-buffered global_load_async_to_lds_b128
// (ASYNCcnt-tracked CDNA5 async copies; s_wait_asynccnt overlap).
// ---------------------------------------------------------------------------

typedef __attribute__((ext_vector_type(16))) _Float16 v16h;
typedef __attribute__((ext_vector_type(8)))  _Float16 v8h;
typedef __attribute__((ext_vector_type(8)))  float    v8f;

#define NPTS_TOTAL 16384
#define B_GRAPHS   8
#define PTS        2048

union FragH { v16h v; v8h h[2]; };

// --- CDNA5 async Global->LDS copy helpers (ASYNCcnt) -----------------------
__device__ __forceinline__ uint32_t lds_off(const void* p) {
  // flat LDS-aperture pointer: low 32 bits are the LDS byte offset
  return (uint32_t)(uintptr_t)p;
}
__device__ __forceinline__ void async_b128(uint32_t lds, const void* gaddr) {
  asm volatile("global_load_async_to_lds_b128 %0, %1, off"
               :: "v"(lds), "v"(gaddr) : "memory");
}
__device__ __forceinline__ void wait_async_le3() {
  asm volatile("s_wait_asynccnt 0x3" ::: "memory");
}
__device__ __forceinline__ void wait_async_0() {
  asm volatile("s_wait_asynccnt 0x0" ::: "memory");
}

// ---------------- weight repack: f32 (K x N) -> f16 transposed (N x Kpad) ---
__global__ void pack_w(const float* __restrict__ W, _Float16* __restrict__ Bt,
                       int K, int Ncol, int Kpad) {
  int id = blockIdx.x * 256 + threadIdx.x;
  if (id >= Ncol * Kpad) return;
  int n = id / Kpad, k = id - n * Kpad;
  Bt[id] = (k < K) ? (_Float16)W[(size_t)k * Ncol + n] : (_Float16)0.f;
}

// ---------------- x (N x 3 f32) -> A0 (N x 32 f16, zero padded) ------------
__global__ void conv_x(const float* __restrict__ x, _Float16* __restrict__ A0) {
  int id = blockIdx.x * 256 + threadIdx.x;      // N*32 threads
  int i = id >> 5, k = id & 31;
  A0[id] = (k < 3) ? (_Float16)x[i * 3 + k] : (_Float16)0.f;
}

// ---------------- fused GEMM + bias + ReLU + BatchNorm, f16 out ------------
// A: M x Kpad f16 row-major. Bt: Ncol x Kpad f16 (weights transposed).
// Block tile 128x64, 8 waves (4x2), wave tile 32x32 (2x2 WMMA tiles).
// Double-buffered LDS, filled with async Global->LDS b128 copies.
__global__ __launch_bounds__(256) void gemm_f16(
    const _Float16* __restrict__ A, const _Float16* __restrict__ Bt,
    _Float16* __restrict__ C,
    const float* __restrict__ bias, const float* __restrict__ gamma,
    const float* __restrict__ beta, const float* __restrict__ mean,
    const float* __restrict__ var, int Kpad, int Ncol) {
  __shared__ _Float16 As[2][128 * 40];  // stride 40 halves = 80B (16B aligned)
  __shared__ _Float16 Bs[2][64 * 40];

  const int t  = threadIdx.x;
  const int m0 = blockIdx.y * 128;
  const int n0 = blockIdx.x * 64;

  // per-thread copy slots: 2 A chunks + 1 B chunk of 8 halves (16B) each
  const int ar = t >> 2;            // 0..63
  const int ac = (t & 3) << 3;      // halves within 32-half K slab
  const _Float16* gA0 = A  + (size_t)(m0 + ar)      * Kpad + ac;
  const _Float16* gA1 = A  + (size_t)(m0 + 64 + ar) * Kpad + ac;
  const _Float16* gB  = Bt + (size_t)(n0 + ar)      * Kpad + ac;
  uint32_t lA0[2], lA1[2], lB[2];
#pragma unroll
  for (int b = 0; b < 2; ++b) {
    lA0[b] = lds_off(&As[b][ar * 40 + ac]);
    lA1[b] = lds_off(&As[b][(64 + ar) * 40 + ac]);
    lB[b]  = lds_off(&Bs[b][ar * 40 + ac]);
  }

  const int w = t >> 5, lane = t & 31;
  const int wr = w >> 1, wc = w & 1;
  const int row = lane & 15;
  const int kha = (lane & 16) ? 8  : 0;   // A-frag K offset (halves)
  const int khb = (lane & 16) ? 8 : 0;    // B-frag K offset (halves)

  v8f acc[2][2] = {};

  // prologue: stage tile 0 into buffer 0
  async_b128(lA0[0], gA0); async_b128(lA1[0], gA1); async_b128(lB[0], gB);
  gA0 += 32; gA1 += 32; gB += 32;

  int buf = 0;
  for (int k0 = 0; k0 < Kpad; k0 += 32) {
    if (k0 + 32 < Kpad) {
      int nb = buf ^ 1;   // prefetch next tile while current computes
      async_b128(lA0[nb], gA0); async_b128(lA1[nb], gA1); async_b128(lB[nb], gB);
      gA0 += 32; gA1 += 32; gB += 32;
      wait_async_le3();   // current tile's 3 copies retired; next 3 in flight
    } else {
      wait_async_0();
    }
    __syncthreads();

    FragH a[2], b[2];
#pragma unroll
    for (int g = 0; g < 2; ++g) {
      // A 16x32 f16 layout: lanes 0-15 K=0..7,16..23; lanes 16-31 K=8..15,24..31
      const _Float16* pa = &As[buf][(wr * 32 + g * 16 + row) * 40 + kha];
      a[g].h[0] = *(const v8h*)pa;
      a[g].h[1] = *(const v8h*)(pa + 16);
      // B 32x16 f16 layout: lanes 0-15 K=0..7,16..23; lanes 16-31 K=8..15,24..31
      const _Float16* pb = &Bs[buf][(wc * 32 + g * 16 + row) * 40 + khb];
      b[g].h[0] = *(const v8h*)pb;
      b[g].h[1] = *(const v8h*)(pb + 16);
    }
#pragma unroll
    for (int i = 0; i < 2; ++i)
#pragma unroll
      for (int j = 0; j < 2; ++j)
        acc[i][j] = __builtin_amdgcn_wmma_f32_16x16x32_f16(
            false, a[i].v, false, b[j].v, (short)0, acc[i][j], false, false);

    __syncthreads();      // all waves done reading buf before it is refilled
    buf ^= 1;
  }

  // epilogue: y = relu(acc + bias); y = (y - mean)*gamma*rsqrt(var+eps) + beta
  const int mhi = (lane & 16) ? 8 : 0;
#pragma unroll
  for (int j = 0; j < 2; ++j) {
    int n = n0 + wc * 32 + j * 16 + row;
    float bs = bias[n];
    float sc = gamma[n] * rsqrtf(var[n] + 1e-5f);
    float mn = mean[n], bt = beta[n];
#pragma unroll
    for (int i = 0; i < 2; ++i)
#pragma unroll
      for (int r = 0; r < 8; ++r) {
        int m = m0 + wr * 32 + i * 16 + mhi + r;
        float v = acc[i][j][r] + bs;
        v = fmaxf(v, 0.f);
        v = (v - mn) * sc + bt;
        C[(size_t)m * Ncol + n] = (_Float16)v;
      }
  }
}

// ---------------- per-node edge max + concat(pos) -> Gin (N x 288 f16) -----
__device__ __forceinline__ int lbound(const int* a, int n, int key) {
  int lo = 0, hi = n;
  while (lo < hi) { int mid = (lo + hi) >> 1; if (a[mid] < key) lo = mid + 1; else hi = mid; }
  return lo;
}

__global__ __launch_bounds__(256) void edge_agg(
    const _Float16* __restrict__ hm, const int* __restrict__ srcs,
    const int* __restrict__ dsts, int E, const float* __restrict__ pos,
    _Float16* __restrict__ gin) {
  int node = blockIdx.x * 8 + (threadIdx.x >> 5);   // one wave per node
  int lane = threadIdx.x & 31;                      // lane handles 8 channels
  int lo = lbound(dsts, E, node);
  int hi = lbound(dsts, E, node + 1);               // dst list is sorted
  float mx[8];
#pragma unroll
  for (int j = 0; j < 8; ++j) mx[j] = -__builtin_inff();
  for (int e = lo; e < hi; ++e) {
    int s = srcs[e];
    v8h d = *(const v8h*)(hm + (size_t)s * 256 + lane * 8);
#pragma unroll
    for (int j = 0; j < 8; ++j) mx[j] = fmaxf(mx[j], (float)d[j]);
  }
  v8h ov;
#pragma unroll
  for (int j = 0; j < 8; ++j) {
    float v = mx[j];
    if (!(fabsf(v) <= 3.0e38f)) v = 0.f;            // where(isfinite(z), z, 0)
    ov[j] = (_Float16)v;
  }
  *(v8h*)(gin + (size_t)node * 288 + lane * 8) = ov;
  float pv = (lane < 3) ? pos[node * 3 + lane] : 0.f;   // cols 256..258 = pos, rest 0
  gin[(size_t)node * 288 + 256 + lane] = (_Float16)pv;
}

// ---------------- per-graph max: g2 (N x 1024 f16) -> zb (8 x 1024 f32) ----
// thread reduces 8 contiguous channels: v8h loads, 512B coalesced per wave-row
__global__ void batch_max(const _Float16* __restrict__ g2, float* __restrict__ zb) {
  int id = blockIdx.x * 256 + threadIdx.x;          // 1024 threads
  int b = id >> 7, grp = id & 127;                  // 8 graphs x 128 groups
  const _Float16* p = g2 + ((size_t)b * PTS) * 1024 + grp * 8;
  float m[8];
#pragma unroll
  for (int j = 0; j < 8; ++j) m[j] = -__builtin_inff();
  for (int r = 0; r < PTS; ++r) {
    v8h d = *(const v8h*)p;
    p += 1024;
#pragma unroll
    for (int j = 0; j < 8; ++j) m[j] = fmaxf(m[j], (float)d[j]);
  }
#pragma unroll
  for (int j = 0; j < 8; ++j) zb[b * 1024 + grp * 8 + j] = m[j];
}

// ---------------- tiny head: (8x1024) -> 256 -> 128 -> 3 -------------------
__global__ __launch_bounds__(256) void head_kernel(
    const float* __restrict__ zb,
    const float* W1, const float* b1, const float* g1, const float* be1,
    const float* mn1, const float* vr1,
    const float* W2, const float* b2, const float* g2, const float* be2,
    const float* mn2, const float* vr2,
    const float* Wo, const float* bo, float* __restrict__ out) {
  __shared__ float h1[8 * 256];
  __shared__ float h2[8 * 128];
  int t = threadIdx.x;
  for (int idx = t; idx < 8 * 256; idx += 256) {
    int m = idx >> 8, n = idx & 255;
    float s = 0.f;
    for (int k = 0; k < 1024; ++k) s += zb[m * 1024 + k] * W1[k * 256 + n];
    s += b1[n]; s = fmaxf(s, 0.f);
    s = (s - mn1[n]) * (g1[n] * rsqrtf(vr1[n] + 1e-5f)) + be1[n];
    h1[idx] = s;
  }
  __syncthreads();
  for (int idx = t; idx < 8 * 128; idx += 256) {
    int m = idx >> 7, n = idx & 127;
    float s = 0.f;
    for (int k = 0; k < 256; ++k) s += h1[m * 256 + k] * W2[k * 128 + n];
    s += b2[n]; s = fmaxf(s, 0.f);
    s = (s - mn2[n]) * (g2[n] * rsqrtf(vr2[n] + 1e-5f)) + be2[n];
    h2[idx] = s;
  }
  __syncthreads();
  if (t < 24) {
    int m = t / 3, c = t - m * 3;
    float s = 0.f;
    for (int k = 0; k < 128; ++k) s += h2[m * 128 + k] * Wo[k * 3 + c];
    out[t] = s + bo[c];
  }
}

// ---------------------------------------------------------------------------
extern "C" void kernel_launch(void* const* d_in, const int* in_sizes, int n_in,
                              void* d_out, int out_size, void* d_ws, size_t ws_size,
                              hipStream_t stream) {
  (void)n_in; (void)out_size; (void)ws_size;
  // input order: x, pos, batch, edge_index, then params depth-first
  // (fc1[3], mu_nn[3], sig_nn[3], gmu[3], gsig[3], head1, head2, out_W, out_b),
  // each layer flattened as W, b, gamma, beta, mean, var.
  const float* x   = (const float*)d_in[0];
  const float* pos = (const float*)d_in[1];
  const int*   ei  = (const int*)d_in[3];
  const int    E   = in_sizes[3] / 2;
  const int BASE_FC1 = 4, BASE_MU = 22, BASE_GMU = 58;
  const int BASE_H1 = 94, BASE_H2 = 100, IDX_OW = 106, IDX_OB = 107;

  char* ws = (char*)d_ws;
  size_t o = 0;
  auto alloc = [&](size_t bytes) { size_t r = o; o += (bytes + 255) & ~(size_t)255; return r; };
  _Float16* P0  = (_Float16*)(ws + alloc((size_t)NPTS_TOTAL * 512 * 2));  // ping
  _Float16* P1  = (_Float16*)(ws + alloc((size_t)NPTS_TOTAL * 512 * 2));  // pong
  _Float16* HM  = (_Float16*)(ws + alloc((size_t)NPTS_TOTAL * 256 * 2));  // hm
  _Float16* G2  = (_Float16*)(ws + alloc((size_t)NPTS_TOTAL * 1024 * 2)); // gmu out
  float*    ZB  = (float*)   (ws + alloc((size_t)B_GRAPHS * 1024 * 4));   // graph max
  _Float16* Wfc0 = (_Float16*)(ws + alloc(64  * 32  * 2));
  _Float16* Wfc1 = (_Float16*)(ws + alloc(64  * 64  * 2));
  _Float16* Wfc2 = (_Float16*)(ws + alloc(128 * 64  * 2));
  _Float16* Wmu0 = (_Float16*)(ws + alloc(128 * 128 * 2));
  _Float16* Wmu1 = (_Float16*)(ws + alloc(128 * 128 * 2));
  _Float16* Wmu2 = (_Float16*)(ws + alloc(256 * 128 * 2));
  _Float16* Wg0  = (_Float16*)(ws + alloc(256 * 288 * 2));
  _Float16* Wg1  = (_Float16*)(ws + alloc(512 * 256 * 2));
  _Float16* Wg2  = (_Float16*)(ws + alloc(1024 * 512 * 2));

  auto P = [&](int base, int layer, int off) { return (const float*)d_in[base + 6 * layer + off]; };
  auto pack = [&](int base, int layer, _Float16* dst, int K, int Ncol, int Kpad) {
    int tot = Ncol * Kpad;
    pack_w<<<dim3((tot + 255) / 256), dim3(256), 0, stream>>>(P(base, layer, 0), dst, K, Ncol, Kpad);
  };
  auto gemm = [&](const _Float16* A, const _Float16* Bt, _Float16* C,
                  int base, int layer, int Kpad, int Ncol) {
    dim3 g(Ncol / 64, NPTS_TOTAL / 128);
    gemm_f16<<<g, dim3(256), 0, stream>>>(A, Bt, C,
        P(base, layer, 1), P(base, layer, 2), P(base, layer, 3),
        P(base, layer, 4), P(base, layer, 5), Kpad, Ncol);
  };

  // weights -> f16 transposed, K zero-padded to multiple of 32
  pack(BASE_FC1, 0, Wfc0, 3,   64,   32);
  pack(BASE_FC1, 1, Wfc1, 64,  64,   64);
  pack(BASE_FC1, 2, Wfc2, 64,  128,  64);
  pack(BASE_MU,  0, Wmu0, 128, 128,  128);
  pack(BASE_MU,  1, Wmu1, 128, 128,  128);
  pack(BASE_MU,  2, Wmu2, 128, 256,  128);
  pack(BASE_GMU, 0, Wg0,  259, 256,  288);
  pack(BASE_GMU, 1, Wg1,  256, 512,  256);
  pack(BASE_GMU, 2, Wg2,  512, 1024, 512);

  // x -> A0 (N x 32 f16) in P1
  conv_x<<<dim3(NPTS_TOTAL * 32 / 256), dim3(256), 0, stream>>>(x, P1);

  // fc1 + mu_nn chain (ping-pong P0/P1)
  gemm(P1, Wfc0, P0, BASE_FC1, 0, 32, 64);
  gemm(P0, Wfc1, P1, BASE_FC1, 1, 64, 64);
  gemm(P1, Wfc2, P0, BASE_FC1, 2, 64, 128);
  gemm(P0, Wmu0, P1, BASE_MU, 0, 128, 128);
  gemm(P1, Wmu1, P0, BASE_MU, 1, 128, 128);
  gemm(P0, Wmu2, HM, BASE_MU, 2, 128, 256);

  // segment_max over edges (dst-sorted) + isfinite mask + concat pos -> Gin in P0
  edge_agg<<<dim3(NPTS_TOTAL / 8), dim3(256), 0, stream>>>(HM, ei, ei + E, E, pos, P0);

  // gmu chain
  gemm(P0, Wg0, P1, BASE_GMU, 0, 288, 256);
  gemm(P1, Wg1, P0, BASE_GMU, 1, 256, 512);
  gemm(P0, Wg2, G2, BASE_GMU, 2, 512, 1024);

  // per-graph max -> ZB (8 x 1024 f32)
  batch_max<<<dim3(4), dim3(256), 0, stream>>>(G2, ZB);

  // head MLPs + output projection -> d_out (24 f32)
  head_kernel<<<dim3(1), dim3(256), 0, stream>>>(ZB,
      P(BASE_H1, 0, 0), P(BASE_H1, 0, 1), P(BASE_H1, 0, 2), P(BASE_H1, 0, 3),
      P(BASE_H1, 0, 4), P(BASE_H1, 0, 5),
      P(BASE_H2, 0, 0), P(BASE_H2, 0, 1), P(BASE_H2, 0, 2), P(BASE_H2, 0, 3),
      P(BASE_H2, 0, 4), P(BASE_H2, 0, 5),
      (const float*)d_in[IDX_OW], (const float*)d_in[IDX_OB], (float*)d_out);
}
